// SS2D_23519240913610
// MI455X (gfx1250) — matrix operand order repaired and checked by reading, not compiled, not run on hardware
//
#include <hip/hip_runtime.h>
#include <stdint.h>

#define BDIM 2
#define HDIM 64
#define WDIM 64
#define LLEN 4096
#define DMODEL 128
#define DINNER 256
#define NSTATE 16
#define RRANK 8
#define KDIR 4
#define CDBL 40   /* RRANK + 2*NSTATE */

typedef __attribute__((ext_vector_type(16))) _Float16 v16h;
typedef __attribute__((ext_vector_type(8)))  float    v8f;

static __device__ __forceinline__ float silu_f(float v) {
  return v / (1.0f + __expf(-v));
}

// A fragment (16x32 f16, M x K), per ISA 7.12.2: lanes 0-15 hold rows M=0..15,
// halves [0..7]=K(kk+hi*8 .. +7), halves [8..15]=K(kk+16+hi*8 ..). Source is f32, cvt on load.
static __device__ __forceinline__ v16h load_a_frag_f32(const float* __restrict__ A, int lda,
                                                       int row0, int kk, int lane) {
  const int r  = row0 + (lane & 15);
  const int hi = lane >> 4;
  const float* p0 = A + (size_t)r * lda + kk + hi * 8;
  const float* p1 = p0 + 16;
  v16h a;
#pragma unroll
  for (int j = 0; j < 8; ++j) {
    a[j]     = (_Float16)p0[j];
    a[8 + j] = (_Float16)p1[j];
  }
  return a;
}

// B fragment (32x16 f16, K x N) with B[k,n] = Wt[n*ldb + k] (row-major weight [N,K]).
// Lanes 0-15 = cols, halves [0..7] -> K = kk + hb*8 + j, [8..15] -> K = kk + 16 + hb*8 + j. Out-of-range cols -> 0.
static __device__ __forceinline__ v16h load_b_frag_w(const float* __restrict__ Wt, int ldb,
                                                     int col0, int kk, int lane, int nvalid) {
  const int n  = col0 + (lane & 15);
  const int hb = lane >> 4;
  v16h b;
  if (n < nvalid) {
    const float* p = Wt + (size_t)n * ldb + kk + hb * 8;
#pragma unroll
    for (int j = 0; j < 8; ++j) { b[j] = (_Float16)p[j]; b[8 + j] = (_Float16)p[16 + j]; }
  } else {
#pragma unroll
    for (int j = 0; j < 16; ++j) b[j] = (_Float16)0.0f;
  }
  return b;
}

// ---------------- in_proj: xz = x @ w_in.T, split -> xi, silu(z) ----------------
// M = B*L = 8192, N = 512, K = 128. One wave per 16x16 tile.
__global__ __launch_bounds__(256) void k_in_proj(const float* __restrict__ x,
                                                 const float* __restrict__ w_in,
                                                 float* __restrict__ xi,
                                                 float* __restrict__ z_silu) {
  const int lane = threadIdx.x & 31;
  const int wave = threadIdx.x >> 5;
  const int tile = blockIdx.x * 8 + wave;   // 16384 tiles
  const int tm = tile & 511;
  const int tn = tile >> 9;                 // 0..31
  v8f acc = {};
#pragma unroll
  for (int kk = 0; kk < DMODEL; kk += 32) {
    v16h a = load_a_frag_f32(x, DMODEL, tm * 16, kk, lane);
    v16h b = load_b_frag_w(w_in, DMODEL, tn * 16, kk, lane, 2 * DINNER);
    acc = __builtin_amdgcn_wmma_f32_16x16x32_f16(false, a, false, b, (short)0, acc, false, false);
  }
  const int n = tn * 16 + (lane & 15);
  const int rbase = tm * 16 + (lane >> 4) * 8;
#pragma unroll
  for (int i = 0; i < 8; ++i) {
    const int m = rbase + i;
    const float v = acc[i];
    if (n < DINNER) xi[(size_t)m * DINNER + n] = v;
    else            z_silu[(size_t)m * DINNER + (n - DINNER)] = silu_f(v);
  }
}

// ---------------- depthwise 3x3 conv + bias + silu ----------------
__global__ __launch_bounds__(256) void k_dwconv(const float* __restrict__ xi,
                                                const float* __restrict__ cw,
                                                const float* __restrict__ cb,
                                                float* __restrict__ xcT) {
  const int d = threadIdx.x;
  const int pix = blockIdx.x;          // 0..B*L-1
  const int b = pix >> 12;
  const int l = pix & (LLEN - 1);
  const int h = l >> 6, w = l & 63;
  float acc = cb[d];
#pragma unroll
  for (int dy = -1; dy <= 1; ++dy) {
#pragma unroll
    for (int dx = -1; dx <= 1; ++dx) {
      const int hh = h + dy, ww = w + dx;
      if (hh >= 0 && hh < HDIM && ww >= 0 && ww < WDIM) {
        acc += xi[((size_t)(b * LLEN + hh * WDIM + ww)) * DINNER + d] *
               cw[d * 9 + (dy + 1) * 3 + (dx + 1)];
      }
    }
  }
  xcT[(size_t)pix * DINNER + d] = silu_f(acc);
}

// ---------------- x_proj: xdbl[b,k][l, c] = sum_d xcT[b][l,d] * xpw[k][c,d] ----------------
__global__ __launch_bounds__(256) void k_xproj(const float* __restrict__ xcT,
                                               const float* __restrict__ xpw,
                                               float* __restrict__ xdbl) {
  const int lane = threadIdx.x & 31;
  const int wave = threadIdx.x >> 5;
  const int tile = blockIdx.x * 8 + wave;   // 6144 tiles
  const int bk = tile / 768;
  const int rem = tile - bk * 768;
  const int tn = rem >> 8;                  // 0..2
  const int tm = rem & 255;                 // 0..255
  const int b = bk >> 2, k = bk & 3;
  const float* A  = xcT + (size_t)b * LLEN * DINNER;
  const float* Wt = xpw + (size_t)k * CDBL * DINNER;
  v8f acc = {};
#pragma unroll
  for (int kk = 0; kk < DINNER; kk += 32) {
    v16h a   = load_a_frag_f32(A, DINNER, tm * 16, kk, lane);
    v16h bfr = load_b_frag_w(Wt, DINNER, tn * 16, kk, lane, CDBL);
    acc = __builtin_amdgcn_wmma_f32_16x16x32_f16(false, a, false, bfr, (short)0, acc, false, false);
  }
  const int n = tn * 16 + (lane & 15);
  const int rbase = tm * 16 + (lane >> 4) * 8;
  if (n < CDBL) {
#pragma unroll
    for (int i = 0; i < 8; ++i)
      xdbl[((size_t)bk * LLEN + rbase + i) * CDBL + n] = acc[i];
  }
}

// physical column for logical step t of scan direction k (H=W=64)
static __device__ __forceinline__ int perm_idx(int k, int t) {
  int tt = (k >= 2) ? (LLEN - 1 - t) : t;
  if (k & 1) tt = ((tt & 63) << 6) | (tt >> 6);
  return tt;
}

// ---------------- selective scan: 1 block per (b,k), 1 lane per channel d ----------------
// [dt|B|C] chunk staging uses double-buffered global_load_async_to_lds_b128 (ASYNCcnt),
// per-lane x values are register-prefetched per chunk, so the 16-step inner loop is
// pure LDS-broadcast + VALU/TRANS work.
__global__ __launch_bounds__(256) void k_scan(const float* __restrict__ xcT,
                                              const float* __restrict__ xdbl,
                                              const float* __restrict__ dt_w,
                                              const float* __restrict__ dt_b,
                                              const float* __restrict__ A_logs,
                                              const float* __restrict__ Ds,
                                              float* __restrict__ ysbuf) {
  const int d  = threadIdx.x;
  const int bk = blockIdx.x;           // 0..7
  const int b  = bk >> 2, k = bk & 3;
  const int kd = k * DINNER + d;
  float Areg[NSTATE], hreg[NSTATE], dtw[RRANK];
#pragma unroll
  for (int n = 0; n < NSTATE; ++n) {
    Areg[n] = -__expf(A_logs[(size_t)kd * NSTATE + n]);
    hreg[n] = 0.0f;
  }
#pragma unroll
  for (int r = 0; r < RRANK; ++r) dtw[r] = dt_w[(size_t)kd * RRANK + r];
  const float dtb = dt_b[kd];
  const float Dd  = Ds[kd];
  const float* xd  = xdbl + (size_t)bk * LLEN * CDBL;
  const float* xcb = xcT  + (size_t)b  * LLEN * DINNER;
  float* yo = ysbuf + (size_t)bk * LLEN * DINNER;

  __shared__ __align__(16) float sdbl[2][16][CDBL];  // double-buffered chunk [dt(8)|B(16)|C(16)]
  const int tid = threadIdx.x;
  const int srow  = tid / 10;          // staging: 160 lanes, one b128 (4 dwords) each
  const int squad = tid - srow * 10;   // 10 quads of 16B per 40-float row

  // stage chunk starting at logical step tc into buffer `buf` (async, 1 instr per wave)
  auto stage = [&](int buf, int tc) {
    if (tid < 160) {
      const int p = perm_idx(k, tc + srow);
      const uint64_t ga  = (uint64_t)(uintptr_t)(xd + (size_t)p * CDBL + squad * 4);
      const uint32_t lds = (uint32_t)(uintptr_t)&sdbl[buf][srow][squad * 4];
      asm volatile("global_load_async_to_lds_b128 %0, %1, off"
                   :: "v"(lds), "v"(ga) : "memory");
    }
  };

  stage(0, 0);
  for (int tc = 0; tc < LLEN; tc += 16) {
    const int cur = (tc >> 4) & 1;
    if (tc + 16 < LLEN) {
      stage(cur ^ 1, tc + 16);
      if (tid < 160) asm volatile("s_wait_asynccnt 0x1" ::: "memory");  // chunk tc landed
    } else {
      if (tid < 160) asm volatile("s_wait_asynccnt 0x0" ::: "memory");
    }
    __syncthreads();                    // publish staged LDS to all waves

    float xv[16];                       // coalesced per-lane x prefetch for this chunk
#pragma unroll
    for (int t = 0; t < 16; ++t)
      xv[t] = xcb[(size_t)perm_idx(k, tc + t) * DINNER + d];

#pragma unroll
    for (int t = 0; t < 16; ++t) {
      const int p = perm_idx(k, tc + t);
      float dt = dtb;
#pragma unroll
      for (int r = 0; r < RRANK; ++r) dt += dtw[r] * sdbl[cur][t][r];
      const float delta = (dt > 20.0f) ? dt : log1pf(__expf(dt));   // softplus
      const float dx = delta * xv[t];
      float y = 0.0f;
#pragma unroll
      for (int n = 0; n < NSTATE; ++n) {
        const float dA = __expf(delta * Areg[n]);
        hreg[n] = dA * hreg[n] + dx * sdbl[cur][t][RRANK + n];
        y += hreg[n] * sdbl[cur][t][RRANK + NSTATE + n];
      }
      yo[(size_t)p * DINNER + d] = y + Dd * xv[t];  // physical position -> merge is a sum
    }
    __syncthreads();                    // compute done before this buffer is re-staged
  }
}

// ---------------- cross-merge + LayerNorm + z gating ----------------
__global__ __launch_bounds__(256) void k_ln(const float* __restrict__ ysbuf,
                                            const float* __restrict__ z_silu,
                                            const float* __restrict__ ln_w,
                                            const float* __restrict__ ln_b,
                                            float* __restrict__ yln) {
  const int d = threadIdx.x;
  const int row = blockIdx.x;          // b*L + l
  const int b = row >> 12;
  const int l = row & (LLEN - 1);
  float v = 0.0f;
#pragma unroll
  for (int k = 0; k < KDIR; ++k)
    v += ysbuf[(((size_t)(b * KDIR + k)) * LLEN + l) * DINNER + d];
  __shared__ float s1[256], s2[256];
  s1[d] = v; s2[d] = v * v;
  __syncthreads();
  for (int off = 128; off > 0; off >>= 1) {
    if (d < off) { s1[d] += s1[d + off]; s2[d] += s2[d + off]; }
    __syncthreads();
  }
  const float mu   = s1[0] * (1.0f / DINNER);
  const float var  = s2[0] * (1.0f / DINNER) - mu * mu;
  const float rstd = rsqrtf(var + 1e-5f);
  yln[(size_t)row * DINNER + d] =
      ((v - mu) * rstd * ln_w[d] + ln_b[d]) * z_silu[(size_t)row * DINNER + d];
}

// ---------------- out_proj: out = yln @ w_out.T ----------------
__global__ __launch_bounds__(256) void k_out_proj(const float* __restrict__ yln,
                                                  const float* __restrict__ w_out,
                                                  float* __restrict__ out) {
  const int lane = threadIdx.x & 31;
  const int wave = threadIdx.x >> 5;
  const int tile = blockIdx.x * 8 + wave;   // 4096 tiles
  const int tm = tile & 511;
  const int tn = tile >> 9;                 // 0..7
  v8f acc = {};
#pragma unroll
  for (int kk = 0; kk < DINNER; kk += 32) {
    v16h a   = load_a_frag_f32(yln, DINNER, tm * 16, kk, lane);
    v16h bfr = load_b_frag_w(w_out, DINNER, tn * 16, kk, lane, DMODEL);
    acc = __builtin_amdgcn_wmma_f32_16x16x32_f16(false, a, false, bfr, (short)0, acc, false, false);
  }
  const int n = tn * 16 + (lane & 15);
  const int rbase = tm * 16 + (lane >> 4) * 8;
#pragma unroll
  for (int i = 0; i < 8; ++i)
    out[(size_t)(rbase + i) * DMODEL + n] = acc[i];
}

extern "C" void kernel_launch(void* const* d_in, const int* in_sizes, int n_in,
                              void* d_out, int out_size, void* d_ws, size_t ws_size,
                              hipStream_t stream) {
  (void)in_sizes; (void)n_in; (void)out_size; (void)ws_size;
  const float* x      = (const float*)d_in[0];
  const float* w_in   = (const float*)d_in[1];
  const float* conv_w = (const float*)d_in[2];
  const float* conv_b = (const float*)d_in[3];
  const float* xpw    = (const float*)d_in[4];
  const float* dt_w   = (const float*)d_in[5];
  const float* dt_b   = (const float*)d_in[6];
  const float* A_logs = (const float*)d_in[7];
  const float* Ds     = (const float*)d_in[8];
  const float* ln_w   = (const float*)d_in[9];
  const float* ln_b   = (const float*)d_in[10];
  const float* w_out  = (const float*)d_in[11];
  float* out = (float*)d_out;

  float* ws = (float*)d_ws;
  const size_t ML = (size_t)BDIM * LLEN;                 // 8192
  float* z_silu = ws; ws += ML * DINNER;                 // 8 MB
  float* xi     = ws; ws += ML * DINNER;                 // 8 MB (reused as yln)
  float* xcT    = ws; ws += ML * DINNER;                 // 8 MB
  float* xdbl   = ws; ws += (size_t)BDIM * KDIR * LLEN * CDBL;    // 5.25 MB
  float* ysbuf  = ws; ws += (size_t)BDIM * KDIR * LLEN * DINNER;  // 33.5 MB
  float* yln = xi;  // xi dead after conv

  k_in_proj <<<2048, 256, 0, stream>>>(x, w_in, xi, z_silu);
  k_dwconv  <<<(unsigned)ML, 256, 0, stream>>>(xi, conv_w, conv_b, xcT);
  k_xproj   <<<768, 256, 0, stream>>>(xcT, xpw, xdbl);
  k_scan    <<<8, 256, 0, stream>>>(xcT, xdbl, dt_w, dt_b, A_logs, Ds, ysbuf);
  k_ln      <<<(unsigned)ML, 256, 0, stream>>>(ysbuf, z_silu, ln_w, ln_b, yln);
  k_out_proj<<<512, 256, 0, stream>>>(yln, w_out, out);
}
